// RGCNConv_42786464203340
// MI455X (gfx1250) — matrix operand order repaired and checked by reading, not compiled, not run on hardware
//
#include <hip/hip_runtime.h>
#include <math.h>

// ---------------- constants (from reference shapes) ----------------
#define C_DIM 128   // channels
#define H_DIM 8     // heads
#define HD    16    // head dim  (C/H)
#define KC    128   // K staging chunk (compile-time)
#define LDW   132   // padded LDS leading dim for transposed W tile

typedef __attribute__((ext_vector_type(16))) _Float16 v16h;
typedef __attribute__((ext_vector_type(2)))  _Float16 h2;
typedef __attribute__((ext_vector_type(8)))  float    v8f;
typedef __attribute__((ext_vector_type(4)))  unsigned u32x4;
typedef __attribute__((ext_vector_type(4)))  int      i32x4;
typedef __attribute__((ext_vector_type(8)))  int      i32x8;

#if defined(__has_builtin)
#if __has_builtin(__builtin_amdgcn_tensor_load_to_lds)
#define HAVE_TDM 1
#endif
#endif

// ---- ordered-uint encoding for float atomicMax ----
__device__ __forceinline__ unsigned f2ord(float f) {
  unsigned u = __float_as_uint(f);
  return (u & 0x80000000u) ? ~u : (u | 0x80000000u);
}
__device__ __forceinline__ float ord2f(unsigned u) {
  return __uint_as_float((u & 0x80000000u) ? (u & 0x7fffffffu) : ~u);
}

// =====================================================================
// WMMA GEMM:  Out[M,Nout] = epilogue( A[M,K] @ W[K,Nout] + bias )
// MODE 0: x+bias   MODE 1: gelu(x+bias)   MODE 2: x+bias+res
//
// Block = 256 threads (8 wave32) computes a 32 x 128 output tile.
// Wave w owns columns [w*16, w*16+16) and BOTH 16-row tiles, so each B
// fragment is reused for two v_wmma_f32_16x16x32_f16.
//
// A chunk (32 x 128 f32) is DMA'd into LDS by the Tensor Data Mover
// (tensor_load_to_lds, D# per ISA 8.3/8.4; OOB rows zero-filled by HW
// since tensor_dim1 = M-row0), completed with s_wait_tensorcnt, then
// pair-converted to f16 (v_cvt_pk_f16_f32 + b32 LDS stores).
// W chunk is staged transposed (Wt[n][k], padded ld) the same packed way.
// All WMMA fragments are contiguous 16B runs -> ds_load_b128.
// =====================================================================
template <int MODE>
__global__ void __launch_bounds__(256)
wmma_gemm(const float* __restrict__ A, const float* __restrict__ W,
          const float* bias, const float* res, float* __restrict__ Out,
          int M, int K, int Nout) {
  __shared__ _Float16 As[32 * KC];       //  8 KB (f16 A tile)
  __shared__ _Float16 Wt[128 * LDW];     // 33 KB (transposed W tile)
#ifdef HAVE_TDM
  __shared__ float    As32[32 * KC];     // 16 KB (TDM landing buffer)
#endif

  const int row0    = blockIdx.x * 32;
  const int colBase = blockIdx.y * 128;  // Nout is a multiple of 128
  const int tid     = threadIdx.x;
  const int wave    = tid >> 5;
  const int lane    = tid & 31;
  const int m       = lane & 15;
  const int hi      = lane >> 4;         // lane group (0 or 1)
  const int nl      = wave * 16 + m;     // column within block tile
  const int n       = colBase + nl;      // global column

  v8f acc0 = {}, acc1 = {};

  for (int kc = 0; kc < K; kc += KC) {
#ifdef HAVE_TDM
    // ---- A chunk: async TDM DMA global->LDS (f32), wave 0 issues ----
    if (wave == 0) {
      const unsigned lds_off = (unsigned)(size_t)(void*)As32;
      const unsigned long long ga =
          (unsigned long long)(const void*)(A + (size_t)row0 * K + kc);
      const unsigned td0 = (unsigned)(K - kc);   // tensor dim0 (cols left)
      const unsigned td1 = (unsigned)(M - row0); // tensor dim1 (rows left)
      u32x4 g0;
      g0[0] = 1u;                                   // count=1 (valid D#)
      g0[1] = lds_off;                              // lds_addr
      g0[2] = (unsigned)(ga & 0xFFFFFFFFu);         // global_addr[31:0]
      g0[3] = (unsigned)((ga >> 32) & 0x01FFFFFFu)  // global_addr[56:32]
            | (2u << 30);                           // type=2 ("image")
      i32x8 g1;
      g1[0] = (int)(2u << 16);                      // data_size=4B
      g1[1] = (int)((td0 & 0xFFFFu) << 16);         // tensor_dim0[15:0]
      g1[2] = (int)((td0 >> 16) | ((td1 & 0xFFFFu) << 16));
      g1[3] = (int)((td1 >> 16) | (128u << 16));    // tile_dim0=128
      g1[4] = (int)32u;                             // tile_dim1=32
      g1[5] = (int)(unsigned)K;                     // tensor_dim0_stride
      g1[6] = 0;
      g1[7] = 0;
      const i32x4 gz = {0, 0, 0, 0};
#if __clang_major__ >= 23
      const i32x8 gz8 = {0, 0, 0, 0, 0, 0, 0, 0};
      __builtin_amdgcn_tensor_load_to_lds(g0, g1, gz, gz, gz8, 0);
#else
      __builtin_amdgcn_tensor_load_to_lds(g0, g1, gz, gz, 0);
#endif
#if __has_builtin(__builtin_amdgcn_s_wait_tensorcnt)
      __builtin_amdgcn_s_wait_tensorcnt(0);
#endif
    }
    __syncthreads();
    // packed f32 -> f16 convert pass (LDS -> LDS)
    for (int i = tid; i < 32 * 64; i += 256) {
      const int r = i >> 6, cp = (i & 63) * 2;
      const float2 v2 = *(const float2*)&As32[r * KC + cp];
      h2 pk; pk[0] = (_Float16)v2.x; pk[1] = (_Float16)v2.y;
      *(h2*)&As[r * KC + cp] = pk;
    }
#else
    // ---- fallback: direct global->LDS f16 staging of A chunk ----
    for (int i = tid; i < 32 * 64; i += 256) {
      const int r = i >> 6, cp = (i & 63) * 2;
      const int gr = row0 + r;
      float2 v2 = make_float2(0.0f, 0.0f);
      if (gr < M) v2 = *(const float2*)&A[(size_t)gr * K + kc + cp];
      h2 pk; pk[0] = (_Float16)v2.x; pk[1] = (_Float16)v2.y;
      *(h2*)&As[r * KC + cp] = pk;
    }
#endif
    // ---- stage W chunk transposed: Wt[n][k], packed pairs along k ----
    for (int i = tid; i < 64 * 128; i += 256) {
      const int kp = i >> 7, nn = i & 127;          // reads coalesced in nn
      const float* wp = &W[(size_t)(kc + 2 * kp) * Nout + colBase + nn];
      h2 pk; pk[0] = (_Float16)wp[0]; pk[1] = (_Float16)wp[Nout];
      *(h2*)&Wt[nn * LDW + 2 * kp] = pk;
    }
    __syncthreads();

#pragma unroll
    for (int k0 = 0; k0 < KC; k0 += 32) {
      v16h b, a0, a1;
      const int ks = k0 + hi * 16;
#pragma unroll
      for (int j = 0; j < 16; ++j) b[j] = Wt[nl * LDW + ks + j];
      const int kb = k0 + hi * 16;
#pragma unroll
      for (int j = 0; j < 8; ++j) {
        a0[j]     = As[m * KC + kb + j];
        a0[8 + j] = As[m * KC + kb + 8 + j];
        a1[j]     = As[(m + 16) * KC + kb + j];
        a1[8 + j] = As[(m + 16) * KC + kb + 8 + j];
      }
      acc0 = __builtin_amdgcn_wmma_f32_16x16x32_f16(false, a0, false, b,
                                                    (short)0, acc0, false, false);
      acc1 = __builtin_amdgcn_wmma_f32_16x16x32_f16(false, a1, false, b,
                                                    (short)0, acc1, false, false);
    }
    __syncthreads();
  }

  // ---- epilogue (specialized at compile time) ----
  const float bv = bias ? bias[n] : 0.0f;
#pragma unroll
  for (int i = 0; i < 8; ++i) {
    const int r0 = row0 + hi * 8 + i;      // tile 0 row
    const int r1 = r0 + 16;                // tile 1 row
    if (r0 < M) {
      float x = acc0[i] + bv;
      if (MODE == 1)      x = 0.5f * x * (1.0f + erff(x * 0.70710678118f));
      else if (MODE == 2) x += res[(size_t)r0 * Nout + n];
      Out[(size_t)r0 * Nout + n] = x;
    }
    if (r1 < M) {
      float x = acc1[i] + bv;
      if (MODE == 1)      x = 0.5f * x * (1.0f + erff(x * 0.70710678118f));
      else if (MODE == 2) x += res[(size_t)r1 * Nout + n];
      Out[(size_t)r1 * Nout + n] = x;
    }
  }
}

// -------------------- per-call workspace init --------------------
__global__ void init_kernel(float* __restrict__ wv, float* __restrict__ den,
                            unsigned* __restrict__ mx, long NC, long NH) {
  long i = (long)blockIdx.x * blockDim.x + threadIdx.x;
  if (i < NC) wv[i] = 0.0f;
  if (i < NH) { den[i] = 0.0f; mx[i] = f2ord(-3.0e38f); }
}

// ---- per-edge attention score + fused segment-max (per head) ----
__global__ void score_max_kernel(const float* __restrict__ q,
                                 const float* __restrict__ k,
                                 const int* __restrict__ src,
                                 const int* __restrict__ dst,
                                 float* __restrict__ score,
                                 unsigned* __restrict__ mx, int E) {
  int t = blockIdx.x * blockDim.x + threadIdx.x;
  if (t >= E * H_DIM) return;
  int e = t >> 3, h = t & 7;
  int d = dst[e];
  const float* qp = q + (size_t)d * C_DIM + h * HD;
  const float* kp = k + (size_t)src[e] * C_DIM + h * HD;
  float acc = 0.0f;
#pragma unroll
  for (int j = 0; j < HD; ++j) acc += qp[j] * kp[j];
  acc *= 0.25f;                               // 1/sqrt(16)
  acc = fminf(10.0f, fmaxf(-10.0f, acc));
  score[t] = acc;
  atomicMax(&mx[(size_t)d * H_DIM + h], f2ord(acc));
}

__global__ void expsum_kernel(const float* __restrict__ score,
                              const int* __restrict__ dst,
                              const unsigned* __restrict__ mx,
                              float* __restrict__ ex,
                              float* __restrict__ den, int E) {
  int t = blockIdx.x * blockDim.x + threadIdx.x;
  if (t >= E * H_DIM) return;
  int e = t >> 3, h = t & 7;
  float mval = ord2f(mx[(size_t)dst[e] * H_DIM + h]);
  float ev = __expf(score[t] - mval);
  ex[t] = ev;
  atomicAdd(&den[(size_t)dst[e] * H_DIM + h], ev);
}

// ---- m = (vt[etype,src] - efeat) * alpha ; scatter-add into wv[dst] ----
__global__ void message_kernel(const float* __restrict__ vt,
                               const float* __restrict__ efeat,
                               const float* __restrict__ ex,
                               const float* __restrict__ den,
                               const int* __restrict__ src,
                               const int* __restrict__ dst,
                               const int* __restrict__ ety,
                               float* __restrict__ wv, long EC, int Nn) {
  long t = (long)blockIdx.x * blockDim.x + threadIdx.x;
  if (t >= EC) return;
  int e = (int)(t >> 7);           // /C_DIM
  int c = (int)(t & 127);
  int h = c >> 4;                  // /HD
  int d = dst[e];
  float alpha = ex[(size_t)e * H_DIM + h] / den[(size_t)d * H_DIM + h];
  float mval = vt[(((size_t)ety[e] * Nn) + src[e]) * C_DIM + c]
             - efeat[(size_t)e * C_DIM + c];
  atomicAdd(&wv[(size_t)d * C_DIM + c], mval * alpha);
}

// ---- LayerNorm, one wave32 per row (4 elems/lane, shfl reductions) ----
__global__ void __launch_bounds__(256)
ln_kernel(const float* __restrict__ x, const float* __restrict__ g,
          const float* __restrict__ b, float* __restrict__ out, int Nn) {
  int row = blockIdx.x * 8 + (threadIdx.x >> 5);
  if (row >= Nn) return;
  int lane = threadIdx.x & 31;
  int c = lane * 4;
  const float4 v = *(const float4*)(x + (size_t)row * C_DIM + c);
  float s = v.x + v.y + v.z + v.w;
#pragma unroll
  for (int o = 16; o > 0; o >>= 1) s += __shfl_xor(s, o);
  float mu = s * (1.0f / 128.0f);
  float d0 = v.x - mu, d1 = v.y - mu, d2 = v.z - mu, d3 = v.w - mu;
  float vs = d0 * d0 + d1 * d1 + d2 * d2 + d3 * d3;
#pragma unroll
  for (int o = 16; o > 0; o >>= 1) vs += __shfl_xor(vs, o);
  float inv = rsqrtf(vs * (1.0f / 128.0f) + 1e-5f);
  float4 o4;
  o4.x = d0 * inv * g[c + 0] + b[c + 0];
  o4.y = d1 * inv * g[c + 1] + b[c + 1];
  o4.z = d2 * inv * g[c + 2] + b[c + 2];
  o4.w = d3 * inv * g[c + 3] + b[c + 3];
  *(float4*)(out + (size_t)row * C_DIM + c) = o4;
}

// =====================================================================
extern "C" void kernel_launch(void* const* d_in, const int* in_sizes, int n_in,
                              void* d_out, int out_size, void* d_ws, size_t ws_size,
                              hipStream_t stream) {
  const float* feat  = (const float*)d_in[0];
  const float* efeat = (const float*)d_in[1];
  const int*   src   = (const int*)d_in[2];
  const int*   dst   = (const int*)d_in[3];
  const int*   ety   = (const int*)d_in[4];
  // d_in[5] = n_head (device scalar); hardcoded H_DIM = 8
  const float* Wq = (const float*)d_in[6];   const float* bq = (const float*)d_in[7];
  const float* Wk = (const float*)d_in[8];   const float* bk = (const float*)d_in[9];
  const float* Wv = (const float*)d_in[10];  const float* bv = (const float*)d_in[11];
  const float* Wrel  = (const float*)d_in[12];
  const float* Wproj = (const float*)d_in[13]; const float* bproj = (const float*)d_in[14];
  const float* WeW   = (const float*)d_in[15]; const float* beW   = (const float*)d_in[16];
  const float* lng   = (const float*)d_in[17]; const float* lnb   = (const float*)d_in[18];
  const float* W1 = (const float*)d_in[19];  const float* b1 = (const float*)d_in[20];
  const float* W2 = (const float*)d_in[21];  const float* b2 = (const float*)d_in[22];

  const int C = C_DIM, H = H_DIM;
  const int Nn = in_sizes[0] / C;
  const int E  = in_sizes[1] / C;
  const int T  = in_sizes[12] / (C * C);

  const size_t NC = (size_t)Nn * C;
  const size_t EH = (size_t)E * H;
  const size_t NH = (size_t)Nn * H;

  // ---------------- workspace layout (floats), with aliasing ----------------
  float* ws    = (float*)d_ws;
  float* q     = ws;                      // NC   (free after score -> hbuf)
  float* k     = q + NC;                  // NC
  float* v     = k + NC;                  // NC
  float* vt    = v + NC;                  // T*NC (free after message -> hid N*4C)
  float* score = vt + (size_t)T * NC;     // EH
  float* ex    = score + EH;              // EH
  float* den   = ex + EH;                 // NH
  float* wv    = den + NH;                // NC
  float* feat2 = wv + NC;                 // NC   (residual, lives to the end)
  unsigned* mx = (unsigned*)(feat2 + NC); // NH
  float* hbuf  = q;                       // alias: LN output
  float* hid   = vt;                      // alias: MLP hidden (N*4C <= T*NC)

  float* outFeat = (float*)d_out;         // N*C
  float* outE    = outFeat + NC;          // E*C

  const dim3 blk(256);
  const int  gN = (Nn + 31) / 32;         // row tiles for node GEMMs
  const int  gE = (E + 31) / 32;          // row tiles for edge GEMM
  const long initN = (long)((NC > NH ? NC : NH) + 255) / 256;

  // 0) zero accumulators, init segment-max
  init_kernel<<<(unsigned)initN, blk, 0, stream>>>(wv, den, mx, (long)NC, (long)NH);

  // 1) q/k/v projections (WMMA)
  wmma_gemm<0><<<dim3(gN, 1), blk, 0, stream>>>(feat, Wq, bq, nullptr, q, Nn, C, C);
  wmma_gemm<0><<<dim3(gN, 1), blk, 0, stream>>>(feat, Wk, bk, nullptr, k, Nn, C, C);
  wmma_gemm<0><<<dim3(gN, 1), blk, 0, stream>>>(feat, Wv, bv, nullptr, v, Nn, C, C);

  // 2) per-relation transform vt[t] = v @ W_rel[t]  (WMMA, T small)
  for (int t = 0; t < T; ++t)
    wmma_gemm<0><<<dim3(gN, 1), blk, 0, stream>>>(v, Wrel + (size_t)t * C * C,
                                                  nullptr, nullptr,
                                                  vt + (size_t)t * NC, Nn, C, C);

  // 3) edge softmax over incoming edges (score fused with segment-max)
  const unsigned gEH = (unsigned)((EH + 255) / 256);
  score_max_kernel<<<gEH, blk, 0, stream>>>(q, k, src, dst, score, mx, E);
  expsum_kernel  <<<gEH, blk, 0, stream>>>(score, dst, mx, ex, den, E);

  // 4) weighted message + scatter-add
  const long EC = (long)E * C;
  message_kernel<<<(unsigned)((EC + 255) / 256), blk, 0, stream>>>(
      vt, efeat, ex, den, src, dst, ety, wv, EC, Nn);

  // 5) output projection + residual, layernorm, MLP (+GELU), residual
  wmma_gemm<2><<<dim3(gN, 1), blk, 0, stream>>>(wv, Wproj, bproj, feat, feat2, Nn, C, C);
  ln_kernel<<<(Nn + 7) / 8, blk, 0, stream>>>(feat2, lng, lnb, hbuf, Nn);
  wmma_gemm<1><<<dim3(gN, 4), blk, 0, stream>>>(hbuf, W1, b1, nullptr, hid, Nn, C, 4 * C);
  wmma_gemm<2><<<dim3(gN, 1), blk, 0, stream>>>(hid, W2, b2, feat2, outFeat, Nn, 4 * C, C);

  // 6) edge feature transform (largest GEMM: E x C x C)
  wmma_gemm<0><<<dim3(gE, 1), blk, 0, stream>>>(efeat, WeW, beW, nullptr, outE, E, C, C);
}
